// InterpreterCaptum_1185410974356
// MI455X (gfx1250) — matrix operand order repaired and checked by reading, not compiled, not run on hardware
//
#include <hip/hip_runtime.h>
#include <hip/hip_bf16.h>

typedef __attribute__((ext_vector_type(16))) __bf16 v16bf;
typedef __attribute__((ext_vector_type(8)))  float  v8f;
typedef __attribute__((ext_vector_type(4)))  unsigned int u32x4;
typedef __attribute__((ext_vector_type(8)))  unsigned int u32x8;

#define LSEQ   6464
#define NSITE  101
#define NCELL  64
#define DM     64
#define DI     128
#define DSTATE 16
#define XDL    48          // padded row stride for xdbl (dt_rank 4 + B 16 + C 16 = 36 -> pad 48)
#define NCHUNK 101
#define CHLEN  64
#define MTILES (LSEQ / 16) // 404 row tiles of 16

__device__ __forceinline__ v8f wmma_bf16(v16bf a, v16bf b, v8f c) {
  // (neg_a, A, neg_b, B, c_mod, C, reuse_a, reuse_b)
  return __builtin_amdgcn_wmma_f32_16x16x32_bf16(false, a, false, b, (short)0, c, false, false);
}

__device__ __forceinline__ float sigmoidf_(float x) { return 1.f / (1.f + __expf(-x)); }
__device__ __forceinline__ float siluf_(float x)    { return x * sigmoidf_(x); }

// ---- TDM: stage W[K x nfull] f32 row-major from global into LDS (one DMA per block) ----
// D# per CDNA5 ISA: group0 {count,lds_addr,global_addr,type=2},
// group1 {wg_mask=0, data_size=4B, tensor_dim0/1, tile_dim0/1, dim0_stride}. 2D tensor ->
// VADDR2/VADDR3 omitted (NULL). Only wave 0 issues; TENSORcnt waited before barrier.
__device__ __forceinline__ void tdm_stage(const float* __restrict__ W, float* sW,
                                          int nfull, int K) {
  if ((threadIdx.x >> 5) == 0) {
    unsigned long long ga = (unsigned long long)(uintptr_t)W;
    unsigned lds = (unsigned)(uintptr_t)sW;   // generic LDS addr truncates to LDS byte offset
    u32x4 g0;
    g0.x = 1u;                                              // count=1, user descriptor
    g0.y = lds;                                             // lds_addr
    g0.z = (unsigned)ga;                                    // global_addr[31:0]
    g0.w = ((unsigned)(ga >> 32) & 0x01FFFFFFu) | (2u << 30); // global_addr[56:32] | type=2
    u32x8 g1;
    g1[0] = 2u << 16;                                       // wg_mask=0, data_size=2 (4 bytes)
    g1[1] = ((unsigned)nfull & 0xFFFFu) << 16;              // tensor_dim0[15:0]
    g1[2] = (((unsigned)nfull >> 16) & 0xFFFFu)             // tensor_dim0[31:16]
          | (((unsigned)K & 0xFFFFu) << 16);                // tensor_dim1[15:0]
    g1[3] = (((unsigned)K >> 16) & 0xFFFFu)                 // tensor_dim1[31:16]
          | (((unsigned)nfull & 0xFFFFu) << 16);            // tile_dim0 = nfull
    g1[4] = ((unsigned)K & 0xFFFFu);                        // tile_dim1 = K, tile_dim2 unused
    g1[5] = (unsigned)nfull;                                // tensor_dim0_stride[31:0]
    g1[6] = 0u;                                             // stride hi / dim1_stride lo
    g1[7] = 0u;
    asm volatile("tensor_load_to_lds %0, %1" :: "s"(g0), "s"(g1) : "memory");
    __builtin_amdgcn_s_wait_tensorcnt(0);
  }
  __syncthreads();
}

// ---- WMMA fragment builders (CDNA5 wave32 layouts, CDNA5 ISA) ----
// 16-bit A 16x32: lanes 0-15 -> M=lane, elem e -> K = (e<8?0:16) + half*8 + (e&7).
// K runs contiguous in groups of 8 -> vectorized b128 loads.
__device__ __forceinline__ v16bf load_a_frag(const float* __restrict__ A, int lda,
                                             int m0, int kbase, int lane, int dir) {
  int row = m0 + (lane & 15);
  if (dir) row = LSEQ - 1 - row;
  int half = lane >> 4;
  const float* p = A + (size_t)row * lda + kbase + half * 8;
  float4 q0 = *(const float4*)(p);
  float4 q1 = *(const float4*)(p + 4);
  float4 q2 = *(const float4*)(p + 16);
  float4 q3 = *(const float4*)(p + 20);
  v16bf a;
  a[0]=(__bf16)q0.x;  a[1]=(__bf16)q0.y;  a[2]=(__bf16)q0.z;  a[3]=(__bf16)q0.w;
  a[4]=(__bf16)q1.x;  a[5]=(__bf16)q1.y;  a[6]=(__bf16)q1.z;  a[7]=(__bf16)q1.w;
  a[8]=(__bf16)q2.x;  a[9]=(__bf16)q2.y;  a[10]=(__bf16)q2.z; a[11]=(__bf16)q2.w;
  a[12]=(__bf16)q3.x; a[13]=(__bf16)q3.y; a[14]=(__bf16)q3.z; a[15]=(__bf16)q3.w;
  return a;
}
// 16-bit B 32x16: lane -> N = lane&15, elem e -> K = (e<8?0:16) + half*8 + (e&7). Read from LDS-staged W.
__device__ __forceinline__ v16bf load_b_frag(const float* W, int nfull,
                                             int kbase, int n0, int lane) {
  int n = n0 + (lane & 15);
  int half = lane >> 4;
  v16bf b;
#pragma unroll
  for (int e = 0; e < 16; ++e) {
    int k = kbase + ((e >> 3) << 4) + half * 8 + (e & 7);
    float v = (n < nfull) ? W[k * nfull + n] : 0.f;
    b[e] = (__bf16)v;
  }
  return b;
}

// ---- embed: x = (concat(CpG,cell,DNA)+pos) @ W_fcc + b_fcc  (L x 64) ----
__global__ void k_embed(const float* __restrict__ DNA, const float* __restrict__ CpG,
                        const float* __restrict__ cel, const float* __restrict__ pos,
                        const float* __restrict__ Wfcc, const float* __restrict__ bfcc,
                        float* __restrict__ xout) {
  extern __shared__ float sW[];                    // 64x64 f32
  tdm_stage(Wfcc, sW, DM, DM);
  int wave = (blockIdx.x * blockDim.x + threadIdx.x) >> 5;
  if (wave >= MTILES) return;                      // wave-uniform: EXEC stays full for WMMA
  int lane = threadIdx.x & 31;
  int m0 = wave * 16;
  int row = m0 + (lane & 15), half = lane >> 4, nl = lane & 15;

  v16bf a[2];
#pragma unroll
  for (int c = 0; c < 2; ++c) {
#pragma unroll
    for (int e = 0; e < 16; ++e) {
      int k = c * 32 + ((e >> 3) << 4) + half * 8 + (e & 7);
      float xv;
      if (k < 16)      xv = CpG[row * 16 + k];
      else if (k < 32) xv = cel[row * 16 + (k - 16)];
      else             xv = DNA[row * 32 + (k - 32)];
      xv += pos[row * 64 + k];
      a[c][e] = (__bf16)xv;
    }
  }
#pragma unroll
  for (int nt = 0; nt < 4; ++nt) {
    v8f acc = {};
    acc = wmma_bf16(a[0], load_b_frag(sW, DM, 0,  nt * 16, lane), acc);
    acc = wmma_bf16(a[1], load_b_frag(sW, DM, 32, nt * 16, lane), acc);
#pragma unroll
    for (int r = 0; r < 8; ++r) {
      int m = m0 + half * 8 + r;
      int n = nt * 16 + nl;
      xout[m * DM + n] = acc[r] + bfcc[n];
    }
  }
}

// ---- generic WMMA GEMM: out[L x ldo] = A[L x K] @ W[K x nfull], W staged via TDM ----
template<int K, int NTILES>
__global__ void k_gemm(const float* __restrict__ A, int lda, int dir,
                       const float* __restrict__ W, int nfull,
                       float* __restrict__ out, int ldo) {
  extern __shared__ float sW[];                    // K x nfull f32
  tdm_stage(W, sW, nfull, K);
  int wave = (blockIdx.x * blockDim.x + threadIdx.x) >> 5;
  if (wave >= MTILES) return;
  int lane = threadIdx.x & 31;
  int m0 = wave * 16, half = lane >> 4, nl = lane & 15;
  constexpr int NC = K / 32;
  v16bf a[NC];
#pragma unroll
  for (int c = 0; c < NC; ++c) a[c] = load_a_frag(A, lda, m0, c * 32, lane, dir);
#pragma unroll
  for (int nt = 0; nt < NTILES; ++nt) {
    v8f acc = {};
#pragma unroll
    for (int c = 0; c < NC; ++c)
      acc = wmma_bf16(a[c], load_b_frag(sW, nfull, c * 32, nt * 16, lane), acc);
#pragma unroll
    for (int r = 0; r < 8; ++r)
      out[(m0 + half * 8 + r) * ldo + nt * 16 + nl] = acc[r];
  }
}

// ---- out_proj: mbsum[flip?] (+)= 0.5 * (ygate @ W_out), W_out staged via TDM ----
__global__ void k_outproj(const float* __restrict__ G, const float* __restrict__ Wout,
                          float* __restrict__ mbs, int dir, int accum) {
  extern __shared__ float sW[];                    // 128x64 f32
  tdm_stage(Wout, sW, DM, DI);
  int wave = (blockIdx.x * blockDim.x + threadIdx.x) >> 5;
  if (wave >= MTILES) return;
  int lane = threadIdx.x & 31;
  int m0 = wave * 16, half = lane >> 4, nl = lane & 15;
  v16bf a[4];
#pragma unroll
  for (int c = 0; c < 4; ++c) a[c] = load_a_frag(G, DI, m0, c * 32, lane, 0);
#pragma unroll
  for (int nt = 0; nt < 4; ++nt) {
    v8f acc = {};
#pragma unroll
    for (int c = 0; c < 4; ++c)
      acc = wmma_bf16(a[c], load_b_frag(sW, DM, c * 32, nt * 16, lane), acc);
#pragma unroll
    for (int r = 0; r < 8; ++r) {
      int m = m0 + half * 8 + r;
      int ro = dir ? (LSEQ - 1 - m) : m;
      float v = 0.5f * acc[r];
      int o = ro * DM + nt * 16 + nl;
      if (accum) mbs[o] += v; else mbs[o] = v;
    }
  }
}

// ---- depthwise causal conv (width 4) + SiLU, in scan order ----
__global__ void k_conv(const float* __restrict__ xz, const float* __restrict__ cw,
                       const float* __restrict__ cb, float* __restrict__ xc) {
  int idx = blockIdx.x * blockDim.x + threadIdx.x;
  if (idx >= LSEQ * DI) return;
  int l = idx / DI, d = idx % DI;
  float s = cb[d];
#pragma unroll
  for (int t = 0; t < 4; ++t) {
    int ls = l - 3 + t;
    if (ls >= 0) s += xz[ls * 256 + d] * cw[d * 4 + t];
  }
  xc[idx] = siluf_(s);
}

// ---- dt = softplus(xdbl[:, :4] @ W_dt + b_dt) ----
__global__ void k_dt(const float* __restrict__ xdb, const float* __restrict__ Wdt,
                     const float* __restrict__ bdt, float* __restrict__ dtb) {
  int idx = blockIdx.x * blockDim.x + threadIdx.x;
  if (idx >= LSEQ * DI) return;
  int l = idx / DI, d = idx % DI;
  float s = bdt[d];
#pragma unroll
  for (int r = 0; r < 4; ++r) s += xdb[l * XDL + r] * Wdt[r * DI + d];
  dtb[idx] = (s > 20.f) ? s : log1pf(__expf(s));
}

// ---- scan phase 1: per-chunk local scan -> (prod a, local h) per (chunk,d,s) ----
__global__ void k_scan_part(const float* __restrict__ dtb, const float* __restrict__ xc,
                            const float* __restrict__ xdb, const float* __restrict__ Alog,
                            float* __restrict__ P, float* __restrict__ H) {
  __shared__ float Bs[CHLEN * DSTATE];
  int chunk = blockIdx.x, d = threadIdx.x;
  for (int t = threadIdx.x; t < CHLEN * DSTATE; t += blockDim.x)
    Bs[t] = xdb[(chunk * CHLEN + (t >> 4)) * XDL + 4 + (t & 15)];
  __syncthreads();
  float Areg[16], Pr[16], Hr[16];
#pragma unroll
  for (int s = 0; s < 16; ++s) { Areg[s] = -__expf(Alog[d * 16 + s]); Pr[s] = 1.f; Hr[s] = 0.f; }
  for (int i = 0; i < CHLEN; ++i) {
    int l = chunk * CHLEN + i;
    if (l + 16 < LSEQ) __builtin_prefetch(&dtb[(l + 16) * DI + d], 0, 1);
    float dtv = dtb[l * DI + d], xv = xc[l * DI + d];
    float bx = dtv * xv;
#pragma unroll
    for (int s = 0; s < 16; ++s) {
      float av = __expf(dtv * Areg[s]);
      Hr[s] = av * Hr[s] + bx * Bs[i * 16 + s];
      Pr[s] *= av;
    }
  }
#pragma unroll
  for (int s = 0; s < 16; ++s) {
    int o = (chunk * DI + d) * 16 + s;
    P[o] = Pr[s]; H[o] = Hr[s];
  }
}

// ---- scan phase 2: sequential combine over 101 chunks per (d,s) ----
__global__ void k_scan_combine(const float* __restrict__ P, const float* __restrict__ H,
                               float* __restrict__ cin) {
  int t = blockIdx.x * blockDim.x + threadIdx.x;   // 0..2047
  float carry = 0.f;
  for (int c = 0; c < NCHUNK; ++c) {
    int o = c * (DI * DSTATE) + t;
    cin[o] = carry;
    carry = P[o] * carry + H[o];
  }
}

// ---- scan phase 3: replay with carry, reduce over states, gate with SiLU(z) ----
__global__ void k_scan_apply(const float* __restrict__ dtb, const float* __restrict__ xc,
                             const float* __restrict__ xdb, const float* __restrict__ Alog,
                             const float* __restrict__ Dres, const float* __restrict__ xz,
                             const float* __restrict__ cin, float* __restrict__ yg) {
  __shared__ float Bs[CHLEN * DSTATE];
  __shared__ float Cs[CHLEN * DSTATE];
  int chunk = blockIdx.x, d = threadIdx.x;
  for (int t = threadIdx.x; t < CHLEN * DSTATE; t += blockDim.x) {
    int base = (chunk * CHLEN + (t >> 4)) * XDL;
    Bs[t] = xdb[base + 4  + (t & 15)];
    Cs[t] = xdb[base + 20 + (t & 15)];
  }
  __syncthreads();
  float Areg[16], Hr[16];
#pragma unroll
  for (int s = 0; s < 16; ++s) {
    Areg[s] = -__expf(Alog[d * 16 + s]);
    Hr[s]   = cin[(chunk * DI + d) * 16 + s];
  }
  float dres = Dres[d];
  for (int i = 0; i < CHLEN; ++i) {
    int l = chunk * CHLEN + i;
    if (l + 16 < LSEQ) __builtin_prefetch(&xz[(l + 16) * 256 + DI + d], 0, 1);
    float dtv = dtb[l * DI + d], xv = xc[l * DI + d];
    float bx = dtv * xv;
    float acc = 0.f;
#pragma unroll
    for (int s = 0; s < 16; ++s) {
      float av = __expf(dtv * Areg[s]);
      Hr[s] = av * Hr[s] + bx * Bs[i * 16 + s];
      acc += Hr[s] * Cs[i * 16 + s];
    }
    float zv = xz[l * 256 + DI + d];
    yg[l * DI + d] = (acc + xv * dres) * siluf_(zv);
  }
}

// ---- residual + LayerNorm + fused (NSITE,NCELL) transpose on output ----
__global__ void k_ln(const float* __restrict__ src, const float* __restrict__ mbs,
                     const float* __restrict__ g, const float* __restrict__ b,
                     float* __restrict__ dst, int mode) {
  int r = (blockIdx.x * blockDim.x + threadIdx.x) >> 5;
  if (r >= LSEQ) return;
  int lane = threadIdx.x & 31;
  float v0 = src[r * DM + lane]      + mbs[r * DM + lane];
  float v1 = src[r * DM + lane + 32] + mbs[r * DM + lane + 32];
  float sum = v0 + v1;
#pragma unroll
  for (int o = 16; o > 0; o >>= 1) sum += __shfl_xor(sum, o, 32);
  float mean = sum * (1.f / 64.f);
  float q = (v0 - mean) * (v0 - mean) + (v1 - mean) * (v1 - mean);
#pragma unroll
  for (int o = 16; o > 0; o >>= 1) q += __shfl_xor(q, o, 32);
  float inv = rsqrtf(q * (1.f / 64.f) + 1e-5f);
  int pr;
  if (mode == 1) { int i = r / NCELL, j = r % NCELL; pr = j * NSITE + i; }   // site -> cell major
  else           { int j = r / NSITE, i = r % NSITE; pr = i * NCELL + j; }   // cell -> site major
  dst[pr * DM + lane]      = (v0 - mean) * inv * g[lane]      + b[lane];
  dst[pr * DM + lane + 32] = (v1 - mean) * inv * g[lane + 32] + b[lane + 32];
}

// ---- head: gather 64 (halfwin,row) positions, dot with W_fc, sigmoid loss ----
__global__ void k_head(const float* __restrict__ x, const float* __restrict__ Wfc,
                       const float* __restrict__ bfc, const float* __restrict__ yt,
                       const int* __restrict__ hw, const int* __restrict__ rows,
                       float* __restrict__ out) {
  int k = threadIdx.x;
  if (k >= 64) return;
  int row = hw[0] * NCELL + rows[k];
  float s = bfc[0];
#pragma unroll
  for (int c = 0; c < DM; ++c) s += x[row * DM + c] * Wfc[c];
  out[k] = 1.f - fabsf(yt[k] - sigmoidf_(s));
}

extern "C" void kernel_launch(void* const* d_in, const int* in_sizes, int n_in,
                              void* d_out, int out_size, void* d_ws, size_t ws_size,
                              hipStream_t stream) {
  (void)in_sizes; (void)n_in; (void)out_size; (void)ws_size;
  const float* DNA  = (const float*)d_in[0];
  const float* CpG  = (const float*)d_in[1];
  const float* cel  = (const float*)d_in[2];
  const float* pos  = (const float*)d_in[3];
  const float* ytru = (const float*)d_in[4];
  const float* Wfcc = (const float*)d_in[5];
  const float* bfcc = (const float*)d_in[6];
  const float* Win  = (const float*)d_in[7];
  const float* cw   = (const float*)d_in[8];
  const float* cb   = (const float*)d_in[9];
  const float* Wxp  = (const float*)d_in[10];
  const float* Wdt  = (const float*)d_in[11];
  const float* bdt  = (const float*)d_in[12];
  const float* Alog = (const float*)d_in[13];
  const float* Dres = (const float*)d_in[14];
  const float* Wout = (const float*)d_in[15];
  const float* lng  = (const float*)d_in[16];
  const float* lnb  = (const float*)d_in[17];
  const float* Wfc  = (const float*)d_in[18];
  const float* bfc  = (const float*)d_in[19];
  const int*   hw   = (const int*)d_in[20];
  const int*   ridx = (const int*)d_in[21];

  float* ws = (float*)d_ws;
  size_t off = 0;
  float* xA   = ws + off; off += (size_t)LSEQ * DM;
  float* xB   = ws + off; off += (size_t)LSEQ * DM;
  float* mbs  = ws + off; off += (size_t)LSEQ * DM;
  float* xz   = ws + off; off += (size_t)LSEQ * 256;
  float* xc   = ws + off; off += (size_t)LSEQ * DI;
  float* xdb  = ws + off; off += (size_t)LSEQ * XDL;
  float* dtb  = ws + off; off += (size_t)LSEQ * DI;
  float* yg   = ws + off; off += (size_t)LSEQ * DI;
  float* Pbuf = ws + off; off += (size_t)NCHUNK * DI * DSTATE;
  float* Hbuf = ws + off; off += (size_t)NCHUNK * DI * DSTATE;
  float* Cin  = ws + off; off += (size_t)NCHUNK * DI * DSTATE;

  const int gTiles = (MTILES + 7) / 8;        // 8 waves (16-row tiles) per 256-thread block
  const int gElem  = (LSEQ * DI + 255) / 256;
  const int gLN    = (LSEQ + 7) / 8;

  k_embed<<<gTiles, 256, DM * DM * 4, stream>>>(DNA, CpG, cel, pos, Wfcc, bfcc, xA);

  float* cur = xA; float* nxt = xB;
  for (int step = 0; step < 8; ++step) {
    for (int dir = 0; dir < 2; ++dir) {
      k_gemm<64, 16><<<gTiles, 256, 64 * 256 * 4, stream>>>(cur, DM, dir, Win, 256, xz, 256);
      k_conv<<<gElem, 256, 0, stream>>>(xz, cw, cb, xc);
      k_gemm<128, 3><<<gTiles, 256, 128 * 36 * 4, stream>>>(xc, DI, 0, Wxp, 36, xdb, XDL);
      k_dt<<<gElem, 256, 0, stream>>>(xdb, Wdt, bdt, dtb);
      k_scan_part<<<NCHUNK, 128, 0, stream>>>(dtb, xc, xdb, Alog, Pbuf, Hbuf);
      k_scan_combine<<<8, 256, 0, stream>>>(Pbuf, Hbuf, Cin);
      k_scan_apply<<<NCHUNK, 128, 0, stream>>>(dtb, xc, xdb, Alog, Dres, xz, Cin, yg);
      k_outproj<<<gTiles, 256, DI * DM * 4, stream>>>(yg, Wout, mbs, dir, dir);
    }
    int mode = (step & 1) ? 2 : 1;
    k_ln<<<gLN, 256, 0, stream>>>(cur, mbs, lng, lnb, nxt, mode);
    float* t = cur; cur = nxt; nxt = t;
  }
  k_head<<<1, 64, 0, stream>>>(cur, Wfc, bfc, ytru, hw, ridx, (float*)d_out);
}
